// GraphConvGNNFused_77326591197422
// MI455X (gfx1250) — compile-verified, matrix operand order checked by reading, not run on hardware
//
#include <hip/hip_runtime.h>
#include <hip/hip_bf16.h>

// ---------------- problem constants (match reference) ----------------
constexpr int NN = 200000;     // nodes per type
constexpr int HH = 128;        // hidden channels
constexpr int NP = 200064;     // NN padded to multiple of 128 (1563 * 128)
constexpr int EE = 600000;     // edges per edge type
constexpr int GG = 64;         // graphs
constexpr int CC = 10;         // classes
constexpr float BN_EPS = 1e-5f;
constexpr int WPITCH = 136;    // LDS weight row pitch in bf16 (128+8) -> conflict-free ds_load_b128

// ---------------- WMMA vector types ----------------
typedef __attribute__((ext_vector_type(8)))  float   v8f;
typedef __attribute__((ext_vector_type(16))) __bf16  v16bf;
typedef __attribute__((ext_vector_type(8)))  __bf16  v8bf;
typedef __attribute__((ext_vector_type(4)))  __bf16  v4bf;

static_assert(sizeof(v8bf) == 16, "v8bf must be 16 bytes (B128 load)");
static_assert(sizeof(v16bf) == 32, "v16bf must be 32 bytes (8 VGPRs)");

union FragBF {
    v16bf v;
    v8bf  h[2];
    __bf16 e[16];
};

// A fragment (16x32 bf16): lane L in [0,15] holds row M=L, K in {kb..kb+7, kb+16..kb+23}
// with kb = kstep*32 + (lane<16 ? 0 : 8).
__device__ __forceinline__ void loadA_f32(FragBF& f, const float* __restrict__ A, int row, int kb) {
    const float* p = A + (long long)row * HH + kb;
    float4 a0 = *(const float4*)(p);
    float4 a1 = *(const float4*)(p + 4);
    float4 a2 = *(const float4*)(p + 16);
    float4 a3 = *(const float4*)(p + 20);
    f.e[0]  = (__bf16)a0.x; f.e[1]  = (__bf16)a0.y; f.e[2]  = (__bf16)a0.z; f.e[3]  = (__bf16)a0.w;
    f.e[4]  = (__bf16)a1.x; f.e[5]  = (__bf16)a1.y; f.e[6]  = (__bf16)a1.z; f.e[7]  = (__bf16)a1.w;
    f.e[8]  = (__bf16)a2.x; f.e[9]  = (__bf16)a2.y; f.e[10] = (__bf16)a2.z; f.e[11] = (__bf16)a2.w;
    f.e[12] = (__bf16)a3.x; f.e[13] = (__bf16)a3.y; f.e[14] = (__bf16)a3.z; f.e[15] = (__bf16)a3.w;
}

__device__ __forceinline__ void loadA_bf(FragBF& f, const __bf16* __restrict__ A, int row, int kb) {
    const __bf16* p = A + (long long)row * HH + kb;
    f.h[0] = *(const v8bf*)(p);
    f.h[1] = *(const v8bf*)(p + 16);
}

// B fragment (32x16 bf16) for B[k][n] = W[n][k], W staged in LDS with row pitch WPITCH.
// lane L: column N = L&15, kb = kstep*32 + (L<16 ? 0 : 8).
__device__ __forceinline__ void loadB_lds(FragBF& f, const __bf16* Wl, int n, int lm, int kb) {
    const __bf16* wr = Wl + (n * 16 + lm) * WPITCH + kb;
    f.h[0] = *(const v8bf*)(wr);
    f.h[1] = *(const v8bf*)(wr + 16);
}

// ---------------- kernels ----------------

__global__ void zero_f32(float* __restrict__ p, long long n) {
    long long i = (long long)blockIdx.x * blockDim.x + threadIdx.x;
    if (i < n) p[i] = 0.f;
}

// fp32 -> bf16 with zero padding to n_total
__global__ void cvt_x(const float* __restrict__ x, __bf16* __restrict__ xb,
                      long long n_valid, long long n_total) {
    long long i = (long long)blockIdx.x * blockDim.x + threadIdx.x;
    if (i < n_total) xb[i] = (i < n_valid) ? (__bf16)x[i] : (__bf16)0.f;
}

// Fuse weights: per layer l produce 5 bf16 128x128 matrices:
//  [Wrel(l,0), Wrel(l,2), Wroot(l,0)+Wroot(l,2), Wrel(l,1), Wroot(l,1)]
// and fused biases biasA[l] = brel(l,0)+brel(l,2), biasB[l] = brel(l,1).
__global__ void prep_weights(const float* __restrict__ Wrel, const float* __restrict__ brel,
                             const float* __restrict__ Wroot, __bf16* __restrict__ Wb,
                             float* __restrict__ biasA, float* __restrict__ biasB) {
    const long long M = 16384;
    long long idx = (long long)blockIdx.x * blockDim.x + threadIdx.x;
    if (idx < 3 * 5 * M) {
        int l = (int)(idx / (5 * M));
        long long r = idx % (5 * M);
        int m = (int)(r / M);
        long long off = r % M;
        float v;
        switch (m) {
            case 0:  v = Wrel[(long long)(l * 3 + 0) * M + off]; break;
            case 1:  v = Wrel[(long long)(l * 3 + 2) * M + off]; break;
            case 2:  v = Wroot[(long long)(l * 3 + 0) * M + off] +
                         Wroot[(long long)(l * 3 + 2) * M + off]; break;
            case 3:  v = Wrel[(long long)(l * 3 + 1) * M + off]; break;
            default: v = Wroot[(long long)(l * 3 + 1) * M + off]; break;
        }
        Wb[idx] = (__bf16)v;
    }
    if (idx < 3 * 128) {
        int l = (int)(idx / 128), c = (int)(idx % 128);
        biasA[idx] = brel[(l * 3 + 0) * 128 + c] + brel[(l * 3 + 2) * 128 + c];
        biasB[idx] = brel[(l * 3 + 1) * 128 + c];
    }
}

// segment_sum: one wave per edge, lane handles 4 channels. bf16 gather, fp32 atomic scatter.
__global__ void scatter_add(const __bf16* __restrict__ xsrc, const int* __restrict__ src,
                            const int* __restrict__ dst, float* __restrict__ agg, int nE) {
    int e = blockIdx.x * 8 + (threadIdx.x >> 5);
    if (e >= nE) return;
    int ln = threadIdx.x & 31;
    int s = src[e], d = dst[e];
    const __bf16* sp = xsrc + (long long)s * HH + ln * 4;
    float* dp = agg + (long long)d * HH + ln * 4;
    v4bf x = *(const v4bf*)sp;
    atomicAdd(dp + 0, (float)x[0]);
    atomicAdd(dp + 1, (float)x[1]);
    atomicAdd(dp + 2, (float)x[2]);
    atomicAdd(dp + 3, (float)x[3]);
}

// Fused GEMM: out = relu( sum_t A_t @ W_t^T + bias ), NF fp32 inputs + 1 bf16 input,
// plus per-channel sum/sumsq accumulation for BatchNorm (training-mode stats).
// Block: 256 threads = 8 waves; each wave computes a 16(M) x 128(N) stripe; K = 128.
// Weights staged once per block into LDS (shared by all 8 waves, conflict-free pitch).
template <int NF>
__global__ __launch_bounds__(256) void gemm_fused(
    const float* __restrict__ Af0, const float* __restrict__ Af1,
    const __bf16* __restrict__ W0, const __bf16* __restrict__ W1,
    const __bf16* __restrict__ Ab, const __bf16* __restrict__ Wr,
    const float* __restrict__ bias, float* __restrict__ out,
    float* __restrict__ gsum, float* __restrict__ gsq) {
    constexpr int NT = NF + 1;                 // total input matrices
    __shared__ __bf16 lds_w[NT * 128 * WPITCH];
    __shared__ float s_sum[HH];
    __shared__ float s_sq[HH];

    int tid = threadIdx.x;
    if (tid < HH) { s_sum[tid] = 0.f; s_sq[tid] = 0.f; }

    // ---- stage weights to LDS: thread t copies row (t/2), half (t%2) of each matrix ----
    {
        const __bf16* Wg[NT];
        Wg[0] = W0;
        if (NF == 2) Wg[1] = W1;
        Wg[NT - 1] = Wr;
        int r = tid >> 1, hf = tid & 1;
#pragma unroll
        for (int t = 0; t < NT; t++) {
            const __bf16* g = Wg[t] + r * 128 + hf * 64;
            __bf16* d = lds_w + t * 128 * WPITCH + r * WPITCH + hf * 64;
#pragma unroll
            for (int q = 0; q < 8; q++)
                *(v8bf*)(d + q * 8) = *(const v8bf*)(g + q * 8);
        }
    }
    __syncthreads();

    int wave = tid >> 5, ln = tid & 31;
    int half = (ln >> 4) & 1;       // lane half selects K sub-block
    int lm = ln & 15;               // row (A) / column (B,D) within tile
    int row = blockIdx.x * 128 + wave * 16 + lm;   // A row this lane streams

    v8f acc[8];
#pragma unroll
    for (int n = 0; n < 8; n++) acc[n] = v8f{0.f, 0.f, 0.f, 0.f, 0.f, 0.f, 0.f, 0.f};

#pragma unroll
    for (int kk = 0; kk < 4; kk++) {
        int kb = kk * 32 + half * 8;
        FragBF a[NT];
        loadA_f32(a[0], Af0, row, kb);
        if (NF == 2) loadA_f32(a[1], Af1, row, kb);
        loadA_bf(a[NT - 1], Ab, row, kb);
#pragma unroll
        for (int t = 0; t < NT; t++) {
            const __bf16* Wl = lds_w + t * 128 * WPITCH;
            FragBF b[8];
#pragma unroll
            for (int n = 0; n < 8; n++) loadB_lds(b[n], Wl, n, lm, kb);
#pragma unroll
            for (int n = 0; n < 8; n++)
                acc[n] = __builtin_amdgcn_wmma_f32_16x16x32_bf16(
                    false, a[t].v, false, b[n].v, (short)0, acc[n], false, false);
        }
    }

    // Epilogue: D layout -> lane column = lm (fixed), row = blockRow + wave*16 + half*8 + j.
#pragma unroll
    for (int n = 0; n < 8; n++) {
        int col = n * 16 + lm;
        float bv = bias[col];
        float ls = 0.f, lq = 0.f;
#pragma unroll
        for (int j = 0; j < 8; j++) {
            int r = blockIdx.x * 128 + wave * 16 + half * 8 + j;
            float v = acc[n][j] + bv;
            v = v > 0.f ? v : 0.f;
            if (r < NN) {
                out[(long long)r * HH + col] = v;
                ls += v;
                lq += v * v;
            }
        }
        atomicAdd(&s_sum[col], ls);
        atomicAdd(&s_sq[col], lq);
    }
    __syncthreads();
    if (tid < HH) {
        atomicAdd(&gsum[tid], s_sum[tid]);
        atomicAdd(&gsq[tid], s_sq[tid]);
    }
}

// Turn accumulated sums into per-channel affine (scale, shift) for both node types.
__global__ void bn_stats(const float* __restrict__ gsum, const float* __restrict__ gsq,
                         const float* __restrict__ gamma, const float* __restrict__ beta,
                         float* __restrict__ scale, float* __restrict__ shift) {
    int t = threadIdx.x;            // 256 = [2 types][128 channels]
    if (t < 256) {
        float invn = 1.f / (float)NN;
        float mu = gsum[t] * invn;
        float var = gsq[t] * invn - mu * mu;
        float rs = rsqrtf(var + BN_EPS);
        float g = gamma[t], b = beta[t];
        scale[t] = g * rs;
        shift[t] = b - g * rs * mu;
    }
}

// Apply BN in place (fp32) and mirror to bf16 for next layer's gather + WMMA A operand.
__global__ void bn_apply(float* __restrict__ y, __bf16* __restrict__ xb,
                         const float* __restrict__ scale, const float* __restrict__ shift,
                         int type) {
    long long i = ((long long)blockIdx.x * blockDim.x + threadIdx.x) * 4;
    if (i >= (long long)NN * HH) return;
    int c = (int)(i & 127);
    const float* sc = scale + type * 128 + c;
    const float* sh = shift + type * 128 + c;
    float4 v = *(float4*)(y + i);
    v.x = sc[0] * v.x + sh[0];
    v.y = sc[1] * v.y + sh[1];
    v.z = sc[2] * v.z + sh[2];
    v.w = sc[3] * v.w + sh[3];
    *(float4*)(y + i) = v;
    v4bf b;
    b[0] = (__bf16)v.x; b[1] = (__bf16)v.y; b[2] = (__bf16)v.z; b[3] = (__bf16)v.w;
    *(v4bf*)(xb + i) = b;
}

// Segment-mean pooling, exploiting sorted batch ids: register accumulation, flush on change.
__global__ void pool_kernel(const float* __restrict__ feats, const int* __restrict__ batch,
                            float* __restrict__ sums, float* __restrict__ cnt) {
    int c = threadIdx.x;                    // 128 channels
    int start = blockIdx.x * 64;
    int end = start + 64; if (end > NN) end = NN;
    if (start >= NN) return;
    float acc = 0.f, cacc = 0.f;
    int cur = batch[start];
    for (int i = start; i < end; i++) {
        int g = batch[i];
        if (g != cur) {
            atomicAdd(&sums[cur * HH + c], acc);
            if (c == 0) atomicAdd(&cnt[cur], cacc);
            acc = 0.f; cacc = 0.f; cur = g;
        }
        acc += feats[(long long)i * HH + c];
        cacc += 1.f;
    }
    atomicAdd(&sums[cur * HH + c], acc);
    if (c == 0) atomicAdd(&cnt[cur], cacc);
}

__global__ void final_linear(const float* __restrict__ sums, const float* __restrict__ cnt,
                             const float* __restrict__ lw, const float* __restrict__ lb,
                             float* __restrict__ out) {
    int t = blockIdx.x * 64 + threadIdx.x;
    if (t >= GG * CC) return;
    int g = t / CC, o = t % CC;
    float invc = 1.f / fmaxf(cnt[g], 1.f);
    float a = lb[o];
    for (int k = 0; k < HH; k++) a += sums[g * HH + k] * invc * lw[o * HH + k];
    out[g * CC + o] = a;
}

// ---------------- host driver ----------------
extern "C" void kernel_launch(void* const* d_in, const int* in_sizes, int n_in,
                              void* d_out, int out_size, void* d_ws, size_t ws_size,
                              hipStream_t stream) {
    const float* x_a    = (const float*)d_in[0];
    const float* x_b    = (const float*)d_in[1];
    const float* Wrel   = (const float*)d_in[2];
    const float* brel   = (const float*)d_in[3];
    const float* Wroot  = (const float*)d_in[4];
    const float* gamma  = (const float*)d_in[5];
    const float* beta   = (const float*)d_in[6];
    const float* linW   = (const float*)d_in[7];
    const float* linb   = (const float*)d_in[8];
    const int* ei_aa    = (const int*)d_in[9];
    const int* ei_ab    = (const int*)d_in[10];
    const int* ei_ba    = (const int*)d_in[11];
    const int* batch_a  = (const int*)d_in[12];
    const int* batch_b  = (const int*)d_in[13];
    float* out = (float*)d_out;

    const long long S = (long long)NP * HH;           // padded per-type activation (floats)
    float* ws   = (float*)d_ws;
    float* agg0 = ws;                                  // a<-a aggregation
    float* agg1 = ws + S;                              // a<-b aggregation
    float* agg2 = ws + 2 * S;                          // b<-a aggregation
    float* ya   = ws + 3 * S;                          // x_a activations fp32
    float* yb   = ws + 4 * S;                          // x_b activations fp32
    __bf16* xab = (__bf16*)(ws + 5 * S);               // x_a bf16 mirror (padded, zeroed)
    __bf16* xbb = xab + S;                             // x_b bf16 mirror
    float* misc = ws + 6 * S;
    __bf16* Wb  = (__bf16*)misc;                       // 3*5 fused bf16 128x128 weights
    float* biasA = misc + 122880;                      // [3][128]
    float* biasB = biasA + 384;                        // [3][128]
    float* gsum  = biasB + 384;                        // [2][128]
    float* gsq   = gsum + 256;                         // [2][128]
    float* scale = gsq + 256;                          // [2][128]
    float* shift = scale + 256;                        // [2][128]
    float* psum  = shift + 256;                        // [G][128]
    float* pcnt  = psum + GG * HH;                     // [G]

    // one-time (per launch) prep: fused bf16 weights + bf16 input mirrors
    prep_weights<<<(3 * 5 * 16384 + 255) / 256, 256, 0, stream>>>(Wrel, brel, Wroot, Wb, biasA, biasB);
    cvt_x<<<(int)((S + 255) / 256), 256, 0, stream>>>(x_a, xab, (long long)NN * HH, S);
    cvt_x<<<(int)((S + 255) / 256), 256, 0, stream>>>(x_b, xbb, (long long)NN * HH, S);

    for (int l = 0; l < 3; l++) {
        zero_f32<<<(int)((3 * S + 255) / 256), 256, 0, stream>>>(agg0, 3 * S);
        zero_f32<<<2, 256, 0, stream>>>(gsum, 512);

        scatter_add<<<(EE + 7) / 8, 256, 0, stream>>>(xab, ei_aa, ei_aa + EE, agg0, EE);
        scatter_add<<<(EE + 7) / 8, 256, 0, stream>>>(xbb, ei_ba, ei_ba + EE, agg1, EE);
        scatter_add<<<(EE + 7) / 8, 256, 0, stream>>>(xab, ei_ab, ei_ab + EE, agg2, EE);

        const __bf16* Wl = Wb + (long long)l * 5 * 16384;
        // out_a = relu(Wrel0^T agg0 + Wrel2^T agg1 + (Wroot0+Wroot2)^T x_a + biasA)
        gemm_fused<2><<<NP / 128, 256, 0, stream>>>(agg0, agg1,
                                                    Wl, Wl + 16384, xab, Wl + 2 * 16384,
                                                    biasA + l * 128, ya, gsum, gsq);
        // out_b = relu(Wrel1^T agg2 + Wroot1^T x_b + biasB)
        gemm_fused<1><<<NP / 128, 256, 0, stream>>>(agg2, nullptr,
                                                    Wl + 3 * 16384, nullptr, xbb, Wl + 4 * 16384,
                                                    biasB + l * 128, yb, gsum + 128, gsq + 128);

        bn_stats<<<1, 256, 0, stream>>>(gsum, gsq, gamma, beta, scale, shift);
        bn_apply<<<(int)(((long long)NN * HH / 4 + 255) / 256), 256, 0, stream>>>(ya, xab, scale, shift, 0);
        bn_apply<<<(int)(((long long)NN * HH / 4 + 255) / 256), 256, 0, stream>>>(yb, xbb, scale, shift, 1);
    }

    zero_f32<<<(GG * HH + GG + 255) / 256, 256, 0, stream>>>(psum, GG * HH + GG);
    pool_kernel<<<(NN + 63) / 64, 128, 0, stream>>>(ya, batch_a, psum, pcnt);
    pool_kernel<<<(NN + 63) / 64, 128, 0, stream>>>(yb, batch_b, psum, pcnt);
    final_linear<<<(GG * CC + 63) / 64, 64, 0, stream>>>(psum, pcnt, linW, linb, out);
}
